// DeltaNetBlock_80556406604633
// MI455X (gfx1250) — matrix operand order repaired and checked by reading, not compiled, not run on hardware
//
#include <hip/hip_runtime.h>
#include <hip/hip_bf16.h>

// ---------------------------------------------------------------------------
// GatedDeltaNet block forward for gfx1250 (MI455X).
// GEMMs run on V_WMMA_F32_16X16X32_BF16 (f32 accumulate) with pre-transposed
// bf16 weights so BOTH operand fragments load as 2x ds_load_b128 per lane.
// Double-buffered LDS, register-staged global loads, 1 barrier per K-tile.
// Delta-rule recurrence: VGPR-resident state, lane-pair shfl reductions,
// 8-timestep LDS staging (2 barriers per 8 steps).
// ---------------------------------------------------------------------------

#define H_   16
#define DK_  64
#define DV_  128

typedef __bf16 bf16;
typedef __attribute__((ext_vector_type(16))) __bf16 v16bf;
typedef __attribute__((ext_vector_type(8)))  __bf16 bf16x8;
typedef __attribute__((ext_vector_type(8)))  float  v8f;

__device__ __forceinline__ bf16 f2bf(float f) {
  unsigned u = __float_as_uint(f);
  u += 0x7FFFu + ((u >> 16) & 1u);           // round-to-nearest-even
  unsigned short s = (unsigned short)(u >> 16);
  return __builtin_bit_cast(bf16, s);
}
__device__ __forceinline__ float bf2f(bf16 b) {
  unsigned short s = __builtin_bit_cast(unsigned short, b);
  return __uint_as_float(((unsigned)s) << 16);
}
__device__ __forceinline__ float silu_f(float x) { return x / (1.f + __expf(-x)); }

__device__ __forceinline__ bf16x8 bf8_zero() {
  uint4 z{0u, 0u, 0u, 0u};
  return __builtin_bit_cast(bf16x8, z);
}

// ---------------------------------------------------------------------------
// Tiled bf16 GEMM:  C(f32, MxN) = A(bf16, MxK) * Bt(bf16, NxK)^T [+ addend]
// Block tile 128x128, BK=64, 8 waves (2M x 4N), wave tile 64x32 (4x2 WMMAs).
// M multiple of 128, K multiple of 64 (true for all calls); N edge tiles are
// zero-filled at staging and guarded at store.
// ---------------------------------------------------------------------------
#define BM 128
#define BN 128
#define BK 64
#define LDK 72   // padded LDS row pitch (144B = 36 dwords) to spread banks

__global__ __launch_bounds__(256)
void gemm_bf16_kernel(const bf16* __restrict__ A,    // M x K row-major
                      const bf16* __restrict__ Bt,   // N x K row-major
                      const float* __restrict__ addend,
                      float* __restrict__ C, int M, int N, int K)
{
  __shared__ __align__(16) bf16 sA[2][BM][LDK];
  __shared__ __align__(16) bf16 sB[2][BN][LDK];

  const int tid  = threadIdx.x;
  const int lane = tid & 31;
  const int wave = tid >> 5;
  const int wm   = wave >> 2;                 // 0..1 : 64 rows
  const int wn   = wave & 3;                  // 0..3 : 32 cols
  const int m0   = blockIdx.y * BM;
  const int n0   = blockIdx.x * BN;

  // staging assignment: 2 threads per row, 32 bf16 each
  const int sr = tid >> 1;                    // 0..127
  const int sc = (tid & 1) * 32;              // 0 or 32

  bf16x8 ra[4], rb[4];
  const bf16* pa_base = A  + (size_t)(m0 + sr) * K + sc;
  const bf16* pb_base = Bt + (size_t)(n0 + sr) * K + sc;
  const bool  b_valid = (n0 + sr) < N;

  auto gload = [&](int k0) {
#pragma unroll
    for (int j = 0; j < 4; ++j) ra[j] = *(const bf16x8*)(pa_base + k0 + 8 * j);
    if (b_valid) {
#pragma unroll
      for (int j = 0; j < 4; ++j) rb[j] = *(const bf16x8*)(pb_base + k0 + 8 * j);
    } else {
#pragma unroll
      for (int j = 0; j < 4; ++j) rb[j] = bf8_zero();
    }
  };
  auto sstore = [&](int buf) {
#pragma unroll
    for (int j = 0; j < 4; ++j) *(bf16x8*)&sA[buf][sr][sc + 8 * j] = ra[j];
#pragma unroll
    for (int j = 0; j < 4; ++j) *(bf16x8*)&sB[buf][sr][sc + 8 * j] = rb[j];
  };

  v8f acc[4][2] = {};

  gload(0);
  sstore(0);
  __syncthreads();
  int buf = 0;

  const int fr = lane & 15;                   // fragment row within 16
  for (int k0 = 0; k0 < K; k0 += BK) {
    const bool more = (k0 + BK) < K;
    if (more) gload(k0 + BK);                 // overlap global with WMMA

#pragma unroll
    for (int kk = 0; kk < BK; kk += 32) {
      // A frag (ISA 7.12.2): lane<16 -> K {0..7,16..23}, lane>=16 -> +8
      const int a_lo = kk + ((lane >> 4) << 3);
      // B frag: same K order as A: lane<16 -> K {0..7,16..23}, lane>=16 -> +8
      const int b_lo = kk + ((lane >> 4) << 3);

      v16bf fa[4], fb[2];
#pragma unroll
      for (int i = 0; i < 4; ++i) {
        const bf16* p = &sA[buf][wm * 64 + i * 16 + fr][0];
        union { v16bf v; bf16x8 h[2]; } u;
        u.h[0] = *(const bf16x8*)(p + a_lo);
        u.h[1] = *(const bf16x8*)(p + a_lo + 16);
        fa[i] = u.v;
      }
#pragma unroll
      for (int j = 0; j < 2; ++j) {
        const bf16* p = &sB[buf][wn * 32 + j * 16 + fr][0];
        union { v16bf v; bf16x8 h[2]; } u;
        u.h[0] = *(const bf16x8*)(p + b_lo);
        u.h[1] = *(const bf16x8*)(p + b_lo + 16);
        fb[j] = u.v;
      }
#pragma unroll
      for (int i = 0; i < 4; ++i)
#pragma unroll
        for (int j = 0; j < 2; ++j)
          acc[i][j] = __builtin_amdgcn_wmma_f32_16x16x32_bf16(
              false, fa[i], false, fb[j], (short)0, acc[i][j], false, false);
    }

    if (more) {
      sstore(buf ^ 1);
      __syncthreads();
      buf ^= 1;
    }
  }

  // C/D layout: VGPR r -> row = r + 8*(lane>=16), col = lane&15
  const int rsel = (lane >> 4) << 3;
  const int csel = lane & 15;
#pragma unroll
  for (int i = 0; i < 4; ++i)
#pragma unroll
    for (int j = 0; j < 2; ++j) {
      const int col = n0 + wn * 32 + j * 16 + csel;
      if (col < N) {
#pragma unroll
        for (int r = 0; r < 8; ++r) {
          const int row = m0 + wm * 64 + i * 16 + rsel + r;
          const size_t idx = (size_t)row * N + col;
          float v = acc[i][j][r];
          if (addend) v += addend[idx];
          C[idx] = v;
        }
      }
    }
}

// ---------------------------------------------------------------------------
// Transpose + convert:  dst(N x K, bf16) = transpose(src(K x N, f32))
// 32x32 LDS tiles, coalesced both directions.
// ---------------------------------------------------------------------------
__global__ __launch_bounds__(256)
void cvtT_f32_bf16_kernel(const float* __restrict__ src, bf16* __restrict__ dst,
                          int K, int N)
{
  __shared__ float tile[32][33];
  const int kb = blockIdx.y * 32;
  const int nb = blockIdx.x * 32;
  const int tx = threadIdx.x & 31;
  const int ty = threadIdx.x >> 5;
#pragma unroll
  for (int i = ty; i < 32; i += 8) {
    const int k = kb + i, n = nb + tx;
    tile[i][tx] = (k < K && n < N) ? src[(size_t)k * N + n] : 0.f;
  }
  __syncthreads();
#pragma unroll
  for (int i = ty; i < 32; i += 8) {
    const int n = nb + i, k = kb + tx;
    if (n < N && k < K) dst[(size_t)n * K + k] = f2bf(tile[tx][i]);
  }
}

// ---------------------------------------------------------------------------
// RMSNorm (mean-square + eps) over `cols`, bf16 output (GEMM A operand)
// ---------------------------------------------------------------------------
__global__ __launch_bounds__(256)
void rmsnorm_bf16_kernel(const float* __restrict__ x, const float* __restrict__ w,
                         bf16* __restrict__ out, int cols)
{
  const int row = blockIdx.x;
  const float* xr = x + (size_t)row * cols;
  float ss = 0.f;
  for (int c = threadIdx.x; c < cols; c += 256) { const float v = xr[c]; ss += v * v; }
#pragma unroll
  for (int o = 16; o > 0; o >>= 1) ss += __shfl_xor(ss, o, 32);
  __shared__ float red[8];
  __shared__ float rinv_s;
  if ((threadIdx.x & 31) == 0) red[threadIdx.x >> 5] = ss;
  __syncthreads();
  if (threadIdx.x == 0) {
    float t = 0.f;
    for (int i = 0; i < 8; ++i) t += red[i];
    rinv_s = rsqrtf(t / (float)cols + 1e-6f);
  }
  __syncthreads();
  const float rinv = rinv_s;
  bf16* orow = out + (size_t)row * cols;
  for (int c = threadIdx.x; c < cols; c += 256)
    orow[c] = f2bf(xr[c] * rinv * w[c]);
}

// beta = sigmoid(h @ b_w);  g = -exp(A_log) * softplus(h @ a_w + dt_bias)
__global__ __launch_bounds__(256)
void beta_g_kernel(const bf16* __restrict__ hB, const float* __restrict__ b_w,
                   const float* __restrict__ a_w, const float* __restrict__ A_log,
                   const float* __restrict__ dt_bias, float* __restrict__ beta,
                   float* __restrict__ g, int D)
{
  const int t = blockIdx.x;
  const int hh = threadIdx.x & 15;
  const int part = threadIdx.x >> 4;
  const bf16* hrow = hB + (size_t)t * D;
  float sb = 0.f, sa = 0.f;
  for (int c = part; c < D; c += 16) {
    const float hv = bf2f(hrow[c]);
    sb += hv * b_w[c * H_ + hh];
    sa += hv * a_w[c * H_ + hh];
  }
  __shared__ float pb[256], pa[256];
  pb[threadIdx.x] = sb; pa[threadIdx.x] = sa;
  __syncthreads();
  if (threadIdx.x < H_) {
    float bsum = 0.f, asum = 0.f;
    for (int p = 0; p < 16; ++p) { bsum += pb[p * 16 + threadIdx.x]; asum += pa[p * 16 + threadIdx.x]; }
    beta[(size_t)t * H_ + threadIdx.x] = 1.f / (1.f + expf(-bsum));
    const float xx = asum + dt_bias[threadIdx.x];
    const float sp = (xx > 20.f) ? xx : log1pf(expf(xx));
    g[(size_t)t * H_ + threadIdx.x] = -expf(A_log[threadIdx.x]) * sp;
  }
}

// depthwise causal conv1d (K=4) + silu  (v path; no l2norm)
__global__ void shortconv_silu_kernel(const float* __restrict__ x, const float* __restrict__ w,
                                      float* __restrict__ out, int T, int C, size_t n)
{
  for (size_t idx = blockIdx.x * 256ULL + threadIdx.x; idx < n; idx += (size_t)gridDim.x * 256ULL) {
    const int c = (int)(idx % C);
    const size_t bt = idx / C;
    const int t = (int)(bt % T);
    const size_t b = bt / T;
    const float* wc = w + (size_t)c * 4;
    float acc = 0.f;
#pragma unroll
    for (int i = 0; i < 4; ++i) {
      const int ti = t - 3 + i;
      if (ti >= 0) acc += x[(b * T + ti) * (size_t)C + c] * wc[i];
    }
    out[idx] = silu_f(acc);
  }
}

// fused conv1d(K=4) + silu + per-head l2norm (q/k paths); one block per
// (b,t,head), 64 threads = one channel each.
__global__ __launch_bounds__(64)
void conv_silu_l2_kernel(const float* __restrict__ x, const float* __restrict__ w,
                         float* __restrict__ out, int T, int C)
{
  const size_t row = blockIdx.x;             // bt*H + h
  const int h = (int)(row % H_);
  const size_t bt = row / H_;
  const int t = (int)(bt % T);
  const size_t b = bt / T;
  const int c = h * 64 + threadIdx.x;
  const float* wc = w + (size_t)c * 4;
  float acc = 0.f;
#pragma unroll
  for (int i = 0; i < 4; ++i) {
    const int ti = t - 3 + i;
    if (ti >= 0) acc += x[(b * T + ti) * (size_t)C + c] * wc[i];
  }
  const float y = silu_f(acc);
  float ss = y * y;
#pragma unroll
  for (int off = 16; off > 0; off >>= 1) ss += __shfl_xor(ss, off, 32);
  __shared__ float s2[2];
  if ((threadIdx.x & 31) == 0) s2[threadIdx.x >> 5] = ss;
  __syncthreads();
  out[row * 64 + threadIdx.x] = y * rsqrtf(s2[0] + s2[1] + 1e-6f);
}

// ---------------------------------------------------------------------------
// Gated delta rule: one block per (b,h); 64x128 fp32 state in VGPRs (32 per
// lane). Lane pair (tid, tid^1) holds the two dk-halves of one dv column, so
// cross-dk reductions are a single shfl_xor(.,1) — no intra-step barriers.
// 8 timesteps staged to LDS per round: 2 barriers per 8 steps.
// ---------------------------------------------------------------------------
#define TCHUNK 8

__global__ __launch_bounds__(256)
void delta_rule_kernel(const float* __restrict__ q, const float* __restrict__ k,
                       const float* __restrict__ v, const float* __restrict__ g,
                       const float* __restrict__ beta, float* __restrict__ o, int T)
{
  const int b  = blockIdx.x / H_;
  const int h  = blockIdx.x % H_;
  const int tid = threadIdx.x;
  const int c  = tid >> 1;                   // dv column 0..127
  const int kg = tid & 1;                    // dk half: 0 or 1

  float S[32];
#pragma unroll
  for (int i = 0; i < 32; ++i) S[i] = 0.f;

  __shared__ float sk[TCHUNK][64], sq[TCHUNK][64], sv[TCHUNK][128];
  __shared__ float sg[TCHUNK], sb[TCHUNK];

  for (int t0 = 0; t0 < T; t0 += TCHUNK) {
    // coalesced staging of 8 timesteps
    for (int i = tid; i < TCHUNK * 64; i += 256) {
      const int tt = i >> 6, e = i & 63;
      const size_t base = (((size_t)b * T + t0 + tt) * H_ + h) * 64;
      sk[tt][e] = k[base + e];
      sq[tt][e] = q[base + e];
    }
    for (int i = tid; i < TCHUNK * 128; i += 256) {
      const int tt = i >> 7, e = i & 127;
      sv[tt][e] = v[(((size_t)b * T + t0 + tt) * H_ + h) * 128 + e];
    }
    if (tid < TCHUNK) {
      const size_t base = ((size_t)b * T + t0 + tid) * H_ + h;
      sg[tid] = expf(g[base]);
      sb[tid] = beta[base];
    }
    __syncthreads();

#pragma unroll
    for (int tt = 0; tt < TCHUNK; ++tt) {
      const float ge = sg[tt], bt = sb[tt];
      const float* kk = &sk[tt][kg * 32];
      const float* qq = &sq[tt][kg * 32];
      float vp = 0.f;
#pragma unroll
      for (int i = 0; i < 32; ++i) { S[i] *= ge; vp += kk[i] * S[i]; }
      vp += __shfl_xor(vp, 1, 32);           // combine dk halves (lane pair)
      const float u = bt * (sv[tt][c] - vp);
      float ov = 0.f;
#pragma unroll
      for (int i = 0; i < 32; ++i) { S[i] += kk[i] * u; ov += qq[i] * S[i]; }
      ov += __shfl_xor(ov, 1, 32);
      if (kg == 0)
        o[(((size_t)b * T + t0 + tt) * H_ + h) * 128 + c] = 0.125f * ov;  // dk^-0.5
    }
    __syncthreads();
  }
}

// per-head gated RMSNorm: bf16( rmsnorm(o) * o_norm_w * silu(gate) )
__global__ __launch_bounds__(128)
void gated_out_norm_kernel(const float* __restrict__ o, const float* __restrict__ gate,
                           const float* __restrict__ onw, bf16* __restrict__ outB)
{
  const size_t row = blockIdx.x;             // BT*H rows of DV=128
  const float v = o[row * 128 + threadIdx.x];
  float ss = v * v;
#pragma unroll
  for (int off = 16; off > 0; off >>= 1) ss += __shfl_xor(ss, off, 32);
  __shared__ float red[4];
  if ((threadIdx.x & 31) == 0) red[threadIdx.x >> 5] = ss;
  __syncthreads();
  const float rinv = rsqrtf((red[0] + red[1] + red[2] + red[3]) / 128.f + 1e-6f);
  const float gt = gate[row * 128 + threadIdx.x];
  outB[row * 128 + threadIdx.x] = f2bf(v * rinv * onw[threadIdx.x] * silu_f(gt));
}

__global__ void silu_mul_bf16_kernel(const float* __restrict__ a, const float* __restrict__ b,
                                     bf16* __restrict__ out, size_t n)
{
  for (size_t i = blockIdx.x * 256ULL + threadIdx.x; i < n; i += (size_t)gridDim.x * 256ULL)
    out[i] = f2bf(silu_f(a[i]) * b[i]);
}

// ---------------------------------------------------------------------------
extern "C" void kernel_launch(void* const* d_in, const int* in_sizes, int n_in,
                              void* d_out, int out_size, void* d_ws, size_t ws_size,
                              hipStream_t stream)
{
  const int Tt = 2048, Dd = 1024, VDd = 2048, FFf = 2752;
  const int Bb = in_sizes[0] / (Tt * Dd);
  const int BT = Bb * Tt;

  const float* hidden  = (const float*)d_in[0];
  const float* norm_w  = (const float*)d_in[1];
  const float* q_w     = (const float*)d_in[2];
  const float* k_w     = (const float*)d_in[3];
  const float* v_w     = (const float*)d_in[4];
  const float* b_w     = (const float*)d_in[5];
  const float* a_w     = (const float*)d_in[6];
  const float* A_log   = (const float*)d_in[7];
  const float* dt_bias = (const float*)d_in[8];
  const float* conv_q  = (const float*)d_in[9];
  const float* conv_k  = (const float*)d_in[10];
  const float* conv_v  = (const float*)d_in[11];
  const float* g_w     = (const float*)d_in[12];
  const float* o_norm  = (const float*)d_in[13];
  const float* o_w     = (const float*)d_in[14];
  const float* pnorm_w = (const float*)d_in[15];
  const float* gate_w  = (const float*)d_in[16];
  const float* up_w    = (const float*)d_in[17];
  const float* down_w  = (const float*)d_in[18];
  float* out = (float*)d_out;

  char* ws = (char*)d_ws;
  size_t off = 0;
  auto alloc = [&](size_t bytes) -> void* {
    off = (off + 255) & ~(size_t)255;
    void* p = ws + off;
    off += bytes;
    return p;
  };

  // transposed bf16 weight copies (N x K layout)
  bf16* qwB    = (bf16*)alloc((size_t)Dd * Dd * 2);
  bf16* kwB    = (bf16*)alloc((size_t)Dd * Dd * 2);
  bf16* vwB    = (bf16*)alloc((size_t)Dd * VDd * 2);
  bf16* gwB    = (bf16*)alloc((size_t)Dd * VDd * 2);
  bf16* owB    = (bf16*)alloc((size_t)VDd * Dd * 2);
  bf16* gatewB = (bf16*)alloc((size_t)Dd * FFf * 2);
  bf16* upwB   = (bf16*)alloc((size_t)Dd * FFf * 2);
  bf16* downwB = (bf16*)alloc((size_t)FFf * Dd * 2);
  // activations
  bf16*  hB    = (bf16*)alloc((size_t)BT * Dd * 2);
  float* qf    = (float*)alloc((size_t)BT * Dd * 4);
  float* kf    = (float*)alloc((size_t)BT * Dd * 4);
  float* vf    = (float*)alloc((size_t)BT * VDd * 4);
  float* gatef = (float*)alloc((size_t)BT * VDd * 4);
  float* betaf = (float*)alloc((size_t)BT * H_ * 4);
  float* gf    = (float*)alloc((size_t)BT * H_ * 4);
  float* q2    = (float*)alloc((size_t)BT * Dd * 4);
  float* k2    = (float*)alloc((size_t)BT * Dd * 4);
  float* v2    = (float*)alloc((size_t)BT * VDd * 4);
  float* of    = (float*)alloc((size_t)BT * VDd * 4);
  bf16*  oB    = (bf16*)alloc((size_t)BT * VDd * 2);
  float* xf    = (float*)alloc((size_t)BT * Dd * 4);
  bf16*  h2B   = (bf16*)alloc((size_t)BT * Dd * 2);
  float* ffg   = (float*)alloc((size_t)BT * FFf * 4);
  float* ffu   = (float*)alloc((size_t)BT * FFf * 4);
  bf16*  ffB   = (bf16*)alloc((size_t)BT * FFf * 2);
  (void)ws_size; (void)n_in; (void)out_size;

  auto cvtT = [&](const float* s, bf16* d, int K, int N) {
    dim3 grid((N + 31) / 32, (K + 31) / 32);
    cvtT_f32_bf16_kernel<<<grid, 256, 0, stream>>>(s, d, K, N);
  };
  auto gemm = [&](const bf16* A, const bf16* Bt, const float* add, float* C,
                  int M, int N, int K) {
    dim3 grid((N + BN - 1) / BN, M / BM);
    gemm_bf16_kernel<<<grid, 256, 0, stream>>>(A, Bt, add, C, M, N, K);
  };

  // weight conversions (fp32 KxN -> bf16 NxK)
  cvtT(q_w,    qwB,    Dd,  Dd);
  cvtT(k_w,    kwB,    Dd,  Dd);
  cvtT(v_w,    vwB,    Dd,  VDd);
  cvtT(g_w,    gwB,    Dd,  VDd);
  cvtT(o_w,    owB,    VDd, Dd);
  cvtT(gate_w, gatewB, Dd,  FFf);
  cvtT(up_w,   upwB,   Dd,  FFf);
  cvtT(down_w, downwB, FFf, Dd);

  // 1) pre-norm
  rmsnorm_bf16_kernel<<<BT, 256, 0, stream>>>(hidden, norm_w, hB, Dd);

  // 2) projections (WMMA GEMMs)
  gemm(hB, qwB, nullptr, qf,    BT, Dd,  Dd);
  gemm(hB, kwB, nullptr, kf,    BT, Dd,  Dd);
  gemm(hB, vwB, nullptr, vf,    BT, VDd, Dd);
  gemm(hB, gwB, nullptr, gatef, BT, VDd, Dd);

  // 3) beta / g
  beta_g_kernel<<<BT, 256, 0, stream>>>(hB, b_w, a_w, A_log, dt_bias, betaf, gf, Dd);

  // 4) short causal conv + silu (+ fused per-head l2norm for q,k)
  conv_silu_l2_kernel<<<BT * H_, 64, 0, stream>>>(qf, conv_q, q2, Tt, Dd);
  conv_silu_l2_kernel<<<BT * H_, 64, 0, stream>>>(kf, conv_k, k2, Tt, Dd);
  shortconv_silu_kernel<<<32768, 256, 0, stream>>>(vf, conv_v, v2, Tt, VDd, (size_t)BT * VDd);

  // 5) gated delta rule recurrence
  delta_rule_kernel<<<Bb * H_, 256, 0, stream>>>(q2, k2, v2, gf, betaf, of, Tt);

  // 6) gated per-head RMSNorm -> bf16 operand for o-projection
  gated_out_norm_kernel<<<BT * H_, 128, 0, stream>>>(of, gatef, o_norm, oB);

  // 7) o-projection + residual
  gemm(oB, owB, hidden, xf, BT, Dd, VDd);

  // 8) post-norm + MLP
  rmsnorm_bf16_kernel<<<BT, 256, 0, stream>>>(xf, pnorm_w, h2B, Dd);
  gemm(h2B, gatewB, nullptr, ffg, BT, FFf, Dd);
  gemm(h2B, upwB,   nullptr, ffu, BT, FFf, Dd);
  silu_mul_bf16_kernel<<<16384, 256, 0, stream>>>(ffg, ffu, ffB, (size_t)BT * FFf);

  // 9) down-projection + residual -> output
  gemm(ffB, downwB, xf, out, BT, Dd, FFf);
}
